// TreeLSTMEncoder_32633161515608
// MI455X (gfx1250) — matrix operand order repaired and checked by reading, not compiled, not run on hardware
//
#include <hip/hip_runtime.h>
#include <hip/hip_bf16.h>

typedef __attribute__((ext_vector_type(16))) __bf16 v16bf;
typedef __attribute__((ext_vector_type(8)))  __bf16 v8bf;
typedef __attribute__((ext_vector_type(8)))  float  v8f;

namespace {
constexpr int B  = 64;
constexpr int L  = 256;
constexpr int C  = 8;
constexpr int D  = 512;
constexpr int M  = 512;
constexpr int NIOU = 3 * M;        // 1536
constexpr int NX   = NIOU + M;     // 2048 (iou || f) combined x-projection width
}

// ---------------------------------------------------------------------------
// WMMA fragment loaders (wave32, bf16 16x16x32, contiguous k order, same for A and B)
// ---------------------------------------------------------------------------
// A (16x32, row-major in memory/LDS): lane r=lane&15 holds row r.
// lanes 0-15 hold K 0..15; lanes 16-31 hold K 16..31.
__device__ __forceinline__ v16bf load_fragA(const __bf16* A, int lda,
                                            int k, int lane) {
  const int r  = lane & 15;
  const int ko = lane & 16;                 // 0 or 16
  const __bf16* p = A + (size_t)r * lda + k + ko;
  v8bf lo = *(const v8bf*)(p);              // K = ko .. ko+7
  v8bf hi = *(const v8bf*)(p + 8);          // K = ko+8 .. ko+15
  return __builtin_shufflevector(lo, hi, 0, 1, 2, 3, 4, 5, 6, 7,
                                         8, 9, 10, 11, 12, 13, 14, 15);
}

// B (32x16): column n=lane&15; lanes 0-15 hold K 0..15, lanes 16-31 hold K 16..31.
// B[k,n] == W[n,k] (we compute A @ W^T), so each lane reads 16 contiguous bf16
// from row n of the row-major weight matrix.
__device__ __forceinline__ v16bf load_fragB(const __bf16* __restrict__ W, int ldw,
                                            int k, int lane) {
  const int n  = lane & 15;
  const int ko = lane & 16;                 // 0 or 16
  return *(const v16bf*)(W + (size_t)n * ldw + k + ko);
}

__device__ __forceinline__ float sigf(float x) {
  return 1.0f / (1.0f + __expf(-x));
}

// ---------------------------------------------------------------------------
// Prep kernels
// ---------------------------------------------------------------------------
__global__ void k_zero_f32(float* __restrict__ p, long n) {
  long i = (long)blockIdx.x * blockDim.x + threadIdx.x;
  const long stride = (long)gridDim.x * blockDim.x;
  for (; i < n; i += stride) p[i] = 0.0f;
}

__global__ void k_f32_to_bf16(const float* __restrict__ s, __bf16* __restrict__ d, long n) {
  long i = (long)blockIdx.x * blockDim.x + threadIdx.x;
  const long stride = (long)gridDim.x * blockDim.x;
  for (; i < n; i += stride) d[i] = (__bf16)s[i];
}

// ---------------------------------------------------------------------------
// Big hoisted x-projection GEMM:
//   Xproj[b*L+t, :] = x[b,t,:] @ [W_ioux;W_fx]^T + (b_ioux+b_iouh || b_fx+b_fh)
// (16384 x 2048) = (16384 x 512) @ (512 x 2048), bf16 in / f32 out.
// One 16x16 tile per wave, 4 waves per block.
// ---------------------------------------------------------------------------
__global__ void k_xproj_gemm(const __bf16* __restrict__ Xbf,
                             const __bf16* __restrict__ Wx,
                             const float* __restrict__ b_ioux,
                             const float* __restrict__ b_iouh,
                             const float* __restrict__ b_fx,
                             const float* __restrict__ b_fh,
                             float* __restrict__ Xproj) {
  const int lane = threadIdx.x & 31;
  const int tn = blockIdx.x * 4 + (threadIdx.x >> 5);  // 0..127 (N tiles)
  const int tm = blockIdx.y;                           // 0..1023 (M tiles)
  const __bf16* A = Xbf + (size_t)tm * 16 * D;
  const __bf16* W = Wx  + (size_t)tn * 16 * D;
  v8f acc = {};
#pragma unroll 4
  for (int k = 0; k < D; k += 32) {
    v16bf a = load_fragA(A, D, k, lane);
    v16bf b = load_fragB(W, D, k, lane);
    acc = __builtin_amdgcn_wmma_f32_16x16x32_bf16(false, a, false, b,
                                                  (short)0, acc, false, false);
  }
  const int n = tn * 16 + (lane & 15);
  const float bias = (n < NIOU) ? (b_ioux[n] + b_iouh[n])
                                : (b_fx[n - NIOU] + b_fh[n - NIOU]);
  float* Cp = Xproj + (size_t)tm * 16 * NX + n;
  const int rb = (lane >> 4) << 3;
#pragma unroll
  for (int j = 0; j < 8; ++j) Cp[(size_t)(j + rb) * NX] = acc[j] + bias;
}

// ---------------------------------------------------------------------------
// Fused per-step kernel: gather children into LDS (bf16) + recurrent GEMMs.
//   P0 (blocks 0..11): iou_acc(64x1536) = child_h_sum(64x512) @ W_iouh^T
//       strip sb=bx/3 (16 b-rows), column chunk bx%3 (32 of 96 col tiles).
//       Gather computes sum_c h*mc into LDS (replicated per chunk; cheap).
//   P1 (blocks 12..43): f_acc(512x512) = c_h(512x512) @ W_fh^T
//       strip sb=bx-12 (16 (b,c)-rows). Gather also stages cs_sc = c_s*mc.
// 8 waves/block; each wave computes a 16x64 tile strip (4 accumulators,
// one LDS A-fragment shared across 4 WMMAs per K-step).
// ---------------------------------------------------------------------------
__global__ void k_step_fused(const float* __restrict__ outputs,
                             const int* __restrict__ x_c,
                             const float* __restrict__ x_m_c,
                             const __bf16* __restrict__ Wh,   // [W_iouh; W_fh]
                             float* __restrict__ iou_acc,
                             float* __restrict__ f_acc,
                             float* __restrict__ cs_sc,
                             int t) {
  __shared__ __bf16 As[16 * M];                       // 16 KB A-strip
  const int bx = blockIdx.x;
  const bool isP0 = bx < 12;

  if (isP0) {
    const int sb = bx / 3;                            // b-strip 0..3
    for (int e = threadIdx.x; e < 16 * M; e += 256) {
      const int r = e >> 9;                           // 0..15
      const int k = e & (M - 1);
      const int b = sb * 16 + r;
      const int*   xc = x_c   + ((size_t)b * L + t) * C;
      const float* xm = x_m_c + ((size_t)b * L + t) * C;
      float sum = 0.0f;
#pragma unroll
      for (int c = 0; c < C; ++c) {
        sum += outputs[((size_t)b * L + xc[c]) * (2 * M) + M + k] * xm[c];
      }
      As[e] = (__bf16)sum;
    }
  } else {
    const int sb = bx - 12;                           // (b,c)-row strip 0..31
    for (int e = threadIdx.x; e < 16 * M; e += 256) {
      const int r = e >> 9;
      const int k = e & (M - 1);
      const int row = sb * 16 + r;                    // row = b*C + c
      const int b = row >> 3, c = row & 7;
      const int   idx = x_c  [((size_t)b * L + t) * C + c];
      const float mc  = x_m_c[((size_t)b * L + t) * C + c];
      const float* src = outputs + ((size_t)b * L + idx) * (2 * M);
      As[e] = (__bf16)src[M + k];                     // child h -> bf16
      cs_sc[(size_t)row * M + k] = src[k] * mc;       // child c_s * mc (f32)
    }
  }
  __syncthreads();

  const int lane = threadIdx.x & 31;
  const int wave = threadIdx.x >> 5;                  // 0..7

  const __bf16* Wbase;
  float* Co;
  int ldc;
  if (isP0) {
    const int sb = bx / 3, chunk = bx % 3;
    const int tn0 = chunk * 32 + wave * 4;            // 0..92
    Wbase = Wh + (size_t)tn0 * 16 * M;
    Co = iou_acc + (size_t)sb * 16 * NIOU + (size_t)tn0 * 16;
    ldc = NIOU;
  } else {
    const int sb = bx - 12;
    const int tn0 = wave * 4;                         // 0..28
    Wbase = Wh + (size_t)NIOU * M + (size_t)tn0 * 16 * M;
    Co = f_acc + (size_t)sb * 16 * M + (size_t)tn0 * 16;
    ldc = M;
  }

  v8f acc0 = {}, acc1 = {}, acc2 = {}, acc3 = {};
#pragma unroll 2
  for (int k = 0; k < M; k += 32) {
    const v16bf a = load_fragA(As, M, k, lane);       // LDS (ds_load_b128)
    acc0 = __builtin_amdgcn_wmma_f32_16x16x32_bf16(false, a, false,
             load_fragB(Wbase + (size_t)0 * 16 * M, M, k, lane), (short)0, acc0, false, false);
    acc1 = __builtin_amdgcn_wmma_f32_16x16x32_bf16(false, a, false,
             load_fragB(Wbase + (size_t)1 * 16 * M, M, k, lane), (short)0, acc1, false, false);
    acc2 = __builtin_amdgcn_wmma_f32_16x16x32_bf16(false, a, false,
             load_fragB(Wbase + (size_t)2 * 16 * M, M, k, lane), (short)0, acc2, false, false);
    acc3 = __builtin_amdgcn_wmma_f32_16x16x32_bf16(false, a, false,
             load_fragB(Wbase + (size_t)3 * 16 * M, M, k, lane), (short)0, acc3, false, false);
  }

  const int n  = lane & 15;
  const int rb = (lane >> 4) << 3;
#pragma unroll
  for (int j = 0; j < 8; ++j) {
    float* rowp = Co + (size_t)(j + rb) * ldc + n;
    rowp[ 0] = acc0[j];
    rowp[16] = acc1[j];
    rowp[32] = acc2[j];
    rowp[48] = acc3[j];
  }
}

// ---------------------------------------------------------------------------
// Per-step gates / cell update / mask blend (f32)
// ---------------------------------------------------------------------------
__global__ void k_step_update(const float* __restrict__ iou_acc,
                              const float* __restrict__ f_acc,
                              const float* __restrict__ cs_sc,
                              const float* __restrict__ Xproj,
                              const float* __restrict__ x_m,
                              float* __restrict__ outputs,
                              float* __restrict__ final_state,
                              int t) {
  const int gid = blockIdx.x * blockDim.x + threadIdx.x;  // 0 .. B*M-1
  const int b = gid >> 9;
  const int m = gid & (M - 1);
  const float* xp = Xproj + ((size_t)b * L + t) * NX;     // biases already folded
  const float iv = sigf(iou_acc[(size_t)b * NIOU + m] + xp[m]);
  const float ov = sigf(iou_acc[(size_t)b * NIOU + M + m] + xp[M + m]);
  const float uv = tanhf(iou_acc[(size_t)b * NIOU + 2 * M + m] + xp[2 * M + m]);
  const float fx = xp[NIOU + m];
  float fsum = 0.0f;
#pragma unroll
  for (int c = 0; c < C; ++c) {
    const size_t o = ((size_t)b * C + c) * M + m;
    fsum += sigf(f_acc[o] + fx) * cs_sc[o];
  }
  const float cv = iv * uv + fsum;
  const float hv = ov * tanhf(cv);
  const float mt = x_m[(size_t)b * L + t];
  float* orow = outputs + ((size_t)b * L + t) * (2 * M);
  const float* prow = outputs + ((size_t)b * L + (t - 1)) * (2 * M);
  const float sc = mt * cv + (1.0f - mt) * prow[m];
  const float sh = mt * hv + (1.0f - mt) * prow[M + m];
  orow[m]     = sc;
  orow[M + m] = sh;
  if (t == L - 1) {
    final_state[(size_t)b * (2 * M) + m]     = sc;
    final_state[(size_t)b * (2 * M) + M + m] = sh;
  }
}

// ---------------------------------------------------------------------------
extern "C" void kernel_launch(void* const* d_in, const int* in_sizes, int n_in,
                              void* d_out, int out_size, void* d_ws, size_t ws_size,
                              hipStream_t stream) {
  (void)in_sizes; (void)n_in; (void)out_size; (void)ws_size;
  const float* x      = (const float*)d_in[0];
  const int*   x_c    = (const int*)  d_in[1];
  const float* x_m    = (const float*)d_in[2];
  const float* x_m_c  = (const float*)d_in[3];
  const float* W_ioux = (const float*)d_in[4];
  const float* b_ioux = (const float*)d_in[5];
  const float* W_iouh = (const float*)d_in[6];
  const float* b_iouh = (const float*)d_in[7];
  const float* W_fx   = (const float*)d_in[8];
  const float* b_fx   = (const float*)d_in[9];
  const float* W_fh   = (const float*)d_in[10];
  const float* b_fh   = (const float*)d_in[11];

  float* out         = (float*)d_out;
  float* outputs     = out;                           // (B, L, 2M)
  float* final_state = out + (size_t)B * L * 2 * M;   // (B, 2M)

  char* ws = (char*)d_ws;
  size_t off = 0;
  auto alloc = [&](size_t bytes) -> void* {
    void* p = ws + off;
    off = (off + bytes + 255) & ~(size_t)255;
    return p;
  };
  __bf16* Wx_bf   = (__bf16*)alloc((size_t)NX * D * sizeof(__bf16));  // [W_ioux;W_fx]
  __bf16* Wh_bf   = (__bf16*)alloc((size_t)NX * M * sizeof(__bf16));  // [W_iouh;W_fh]
  __bf16* x_bf    = (__bf16*)alloc((size_t)B * L * D * sizeof(__bf16));
  float*  Xproj   = (float*) alloc((size_t)B * L * NX * sizeof(float));
  float*  cs_sc   = (float*) alloc((size_t)B * C * M * sizeof(float));
  float*  iou_acc = (float*) alloc((size_t)B * NIOU * sizeof(float));
  float*  f_acc   = (float*) alloc((size_t)B * C * M * sizeof(float));

  // ---- prep (parallel, once) ----
  k_zero_f32<<<2048, 256, 0, stream>>>(out, (long)B * L * 2 * M + (long)B * 2 * M);
  k_f32_to_bf16<<<2048, 256, 0, stream>>>(x, x_bf, (long)B * L * D);
  k_f32_to_bf16<<<512, 256, 0, stream>>>(W_ioux, Wx_bf, (long)NIOU * D);
  k_f32_to_bf16<<<512, 256, 0, stream>>>(W_fx,   Wx_bf + (size_t)NIOU * D, (long)M * D);
  k_f32_to_bf16<<<512, 256, 0, stream>>>(W_iouh, Wh_bf, (long)NIOU * M);
  k_f32_to_bf16<<<512, 256, 0, stream>>>(W_fh,   Wh_bf + (size_t)NIOU * M, (long)M * M);

  // ---- hoisted x-projection GEMM: (16384 x 2048) ----
  dim3 gproj(NX / 16 / 4, (B * L) / 16);   // (32, 1024), 4 waves/block
  k_xproj_gemm<<<gproj, 128, 0, stream>>>(x_bf, Wx_bf, b_ioux, b_iouh, b_fx, b_fh,
                                          Xproj);

  // ---- sequential scan (kernel boundaries = device-wide sync) ----
  for (int t = 1; t < L; ++t) {
    k_step_fused<<<44, 256, 0, stream>>>(outputs, x_c, x_m_c, Wh_bf,
                                         iou_acc, f_acc, cs_sc, t);
    k_step_update<<<(B * M) / 256, 256, 0, stream>>>(iou_acc, f_acc, cs_sc, Xproj,
                                                     x_m, outputs, final_state, t);
  }
}
